// GraphSAGEModel_32916629356788
// MI455X (gfx1250) — matrix operand order repaired and checked by reading, not compiled, not run on hardware
//
#include <hip/hip_runtime.h>
#include <hip/hip_bf16.h>

typedef __attribute__((ext_vector_type(16))) _Float16 v16h;
typedef __attribute__((ext_vector_type(8)))  _Float16 v8h;
typedef __attribute__((ext_vector_type(8)))  float    v8f;

#define DIM 128
#define A_STRIDE 264   // 256 cols + 8-half pad (16B) to rotate LDS banks

// ---------------------------------------------------------------- utilities
__global__ void fill_f32(float* __restrict__ p, long long n, float v) {
    long long i = (long long)blockIdx.x * blockDim.x + threadIdx.x;
    long long stride = (long long)gridDim.x * blockDim.x;
    for (; i < n; i += stride) p[i] = v;
}

__global__ void degree_kernel(const int* __restrict__ dst,
                              float* __restrict__ cnt, int E) {
    int e = blockIdx.x * blockDim.x + threadIdx.x;
    if (e < E) atomicAdd(&cnt[dst[e]], 1.0f);
}

__global__ void inv_kernel(float* __restrict__ cnt, int n) {
    int i = blockIdx.x * blockDim.x + threadIdx.x;
    if (i < n) cnt[i] = 1.0f / fmaxf(cnt[i], 1.0f);
}

// One edge per 32 lanes; each lane moves 4 consecutive floats (float4 gather,
// 4x GLOBAL_ATOMIC_ADD_F32 scatter). x table + accumulator both fit in L2.
__global__ void scatter_kernel(const float* __restrict__ feat,
                               const int* __restrict__ src,
                               const int* __restrict__ dst,
                               float* __restrict__ s, int E) {
    long long gid = (long long)blockIdx.x * blockDim.x + threadIdx.x;
    if (gid >= (long long)E * 32) return;
    int e = (int)(gid >> 5);
    int c = ((int)gid & 31) << 2;
    long long u = src[e];
    long long v = dst[e];
    const float4 val = *(const float4*)(feat + u * DIM + c);
    float* o = s + v * DIM + c;
    atomicAdd(o + 0, val.x);
    atomicAdd(o + 1, val.y);
    atomicAdd(o + 2, val.z);
    atomicAdd(o + 3, val.w);
}

// ------------------------------------------------- fused SAGE layer (WMMA)
// out[m,n] = act( sum_k aggr[m,k]*Wl[n,k] + bl[n] + sum_k x[m,k]*Wr[n,k] )
// Treated as a single K=256 GEMM: A_cat=[aggr | x], B_cat=[Wl^T ; Wr^T].
// Block = 256 threads (8 waves) -> 16 rows x 128 cols of output;
// each wave owns one 16x16 column tile, 8x v_wmma_f32_16x16x32_f16.
// xin may alias out: all reads of xin happen in the LDS staging phase of the
// block that exclusively owns those rows, before any store.
template <bool RELU>
__global__ void sage_gemm_kernel(const float* __restrict__ s,
                                 const float* xin,
                                 const float* __restrict__ inv,
                                 const float* __restrict__ Wl,
                                 const float* __restrict__ bl,
                                 const float* __restrict__ Wr,
                                 float* out, int N) {
    __shared__ _Float16 Ah[16 * A_STRIDE];   // 16 x 256 f16 tile (+pad)

    const int tid = threadIdx.x;
    const int m0  = blockIdx.x * 16;

    // Stage A_cat = [mean_aggr | x] as f16 (coalesced b32 loads, f32->f16 cvt)
    for (int i = tid; i < 16 * 256; i += 256) {
        int row = i >> 8;
        int col = i & 255;
        int r = m0 + row; if (r >= N) r = N - 1;
        float v;
        if (col < DIM) v = s[(long long)r * DIM + col] * inv[r];
        else           v = xin[(long long)r * DIM + (col - DIM)];
        Ah[row * A_STRIDE + col] = (_Float16)v;
    }
    __syncthreads();

    const int wave = tid >> 5;          // 0..7 : output column tile
    const int lane = tid & 31;
    const int mrow = lane & 15;         // A row in tile
    const int half = lane >> 4;         // lane-half selects K sub-range
    const int ncol = (wave << 4) + (lane & 15);   // global output column

    v8f c = {};
    #pragma unroll
    for (int kc = 0; kc < 8; ++kc) {    // K = 256 in chunks of 32
        // --- A fragment: a[0..7]=K[kc*32+half*8 ..], a[8..15]=K[+16]
        const v8h* pa0 = (const v8h*)&Ah[mrow * A_STRIDE + kc * 32 + half * 8];
        const v8h* pa1 = (const v8h*)&Ah[mrow * A_STRIDE + kc * 32 + 16 + half * 8];
        v8h lo = *pa0;
        v8h hi = *pa1;
        v16h a = __builtin_shufflevector(lo, hi,
                 0,1,2,3,4,5,6,7,8,9,10,11,12,13,14,15);

        // --- B fragment: lane holds column ncol, b[0..7]=K[kc*32+half*8 ..], b[8..15]=K[+16].
        // B_cat[k][n] = Wl[n][k] (k<128) else Wr[n][k-128]: row-major reads.
        const float* wrow = (kc < 4)
            ? (Wl + (long long)ncol * DIM + kc * 32 + half * 8)
            : (Wr + (long long)ncol * DIM + (kc - 4) * 32 + half * 8);
        v16h b;
        #pragma unroll
        for (int j = 0; j < 16; ++j) b[j] = (_Float16)wrow[j + (j >= 8 ? 8 : 0)];

        c = __builtin_amdgcn_wmma_f32_16x16x32_f16(
                false, a, false, b, (short)0, c, false, false);
    }

    // Epilogue: bias (+ReLU) and store. VGPR j holds row m0 + j + 8*half.
    const float bias = bl[ncol];
    #pragma unroll
    for (int j = 0; j < 8; ++j) {
        float v = c[j] + bias;
        if (RELU) v = fmaxf(v, 0.0f);
        int r = m0 + j + half * 8;
        if (r < N) out[(long long)r * DIM + ncol] = v;
    }
}

// ------------------------------------------------------------------ driver
extern "C" void kernel_launch(void* const* d_in, const int* in_sizes, int n_in,
                              void* d_out, int out_size, void* d_ws, size_t ws_size,
                              hipStream_t stream) {
    const float* x   = (const float*)d_in[0];
    const float* Wl0 = (const float*)d_in[1];
    const float* bl0 = (const float*)d_in[2];
    const float* Wr0 = (const float*)d_in[3];
    const float* Wl1 = (const float*)d_in[4];
    const float* bl1 = (const float*)d_in[5];
    const float* Wr1 = (const float*)d_in[6];
    const int*   ei  = (const int*)d_in[7];

    const int N = in_sizes[0] / DIM;
    const int E = in_sizes[7] / 2;
    const int* srcIdx = ei;
    const int* dstIdx = ei + E;
    float* out = (float*)d_out;

    // Workspace: [cnt->inv : N floats (padded)] [s : N*DIM floats]
    float* cnt = (float*)d_ws;
    float* s   = cnt + (((long long)N + 255) & ~255ll);
    const long long sElems = (long long)N * DIM;

    const int T = 256;
    const int gridFill   = 2048;
    const int gridEdges  = (E + T - 1) / T;
    const int gridNodes  = (N + T - 1) / T;
    const long long scatterThreads = (long long)E * 32;
    const int gridScatter = (int)((scatterThreads + T - 1) / T);
    const int gridGemm   = (N + 15) / 16;

    // degrees -> inverse mean factor (shared by both layers)
    fill_f32<<<gridFill, T, 0, stream>>>(cnt, N, 0.0f);
    fill_f32<<<gridFill, T, 0, stream>>>(s, sElems, 0.0f);
    degree_kernel<<<gridEdges, T, 0, stream>>>(dstIdx, cnt, E);
    inv_kernel<<<gridNodes, T, 0, stream>>>(cnt, N);

    // Layer 1: scatter x, fused mean+GEMM+bias+ReLU -> d_out (acts as h)
    scatter_kernel<<<gridScatter, T, 0, stream>>>(x, srcIdx, dstIdx, s, E);
    sage_gemm_kernel<true><<<gridGemm, T, 0, stream>>>(
        s, x, cnt, Wl0, bl0, Wr0, out, N);

    // Layer 2: re-zero accumulator, scatter h, fused mean+GEMM+bias -> d_out
    fill_f32<<<gridFill, T, 0, stream>>>(s, sElems, 0.0f);
    scatter_kernel<<<gridScatter, T, 0, stream>>>(out, srcIdx, dstIdx, s, E);
    sage_gemm_kernel<false><<<gridGemm, T, 0, stream>>>(
        s, out, cnt, Wl1, bl1, Wr1, out, N);
}
